// DenseGRU_69363721830571
// MI455X (gfx1250) — compile-verified, matrix operand order checked by reading, not run on hardware
//
#include <hip/hip_runtime.h>

// ---------------------------------------------------------------------------
// DenseGRU for MI455X (gfx1250, wave32, WMMA).
//
// Math note: the reference's softmax is over a singleton axis -> weights == 1,
// so ctx == xt.sum(axis=1) exactly and the attention MLP (W1/W2/V) is dead.
// Remaining work: HW-reduction of x (134 MB, bandwidth bound, streamed NT),
// gi GEMM for all T in parallel (bf16 WMMA, software pipelined), and the
// 32-step sequential gh GEMM + gates in one LDS-resident workgroup.
// ---------------------------------------------------------------------------

typedef __attribute__((ext_vector_type(16))) __bf16    v16bf;
typedef __attribute__((ext_vector_type(8)))  float     v8f;
typedef __attribute__((ext_vector_type(8)))  unsigned  v8u;

#define B_    8
#define HW_   256
#define T_    32
#define C_    512
#define H_    512
#define OUT_  4
#define KIN   544      // C+OUT padded to 17*32 for the input GEMM
#define KBI   17       // K-blocks (of 32) for inp @ W_ih^T
#define KBH   16       // K-blocks for h @ W_hh^T  (512 = 16*32)
#define NB    96       // 1536 / 16 output tiles
#define N3H   1536

// round-to-nearest-even f32 -> bf16, pack two into a dword (K even in [15:0])
__device__ inline unsigned pack_bf16(float a, float b) {
  unsigned ua = __builtin_bit_cast(unsigned, a);
  unsigned ub = __builtin_bit_cast(unsigned, b);
  ua = (ua + 0x7FFFu + ((ua >> 16) & 1u)) >> 16;
  ub = (ub + 0x7FFFu + ((ub >> 16) & 1u)) >> 16;
  return (ub << 16) | (ua & 0xFFFFu);
}

// ISA 16-bit A-matrix 16x32 per-lane K pattern (element e=0; e=1 is +1):
// K = (v/4)*16 + (lane/16)*8 + (v%4)*2
__device__ inline int kpat(int v, int lane) {
  return ((v >> 2) << 4) + ((lane >> 4) << 3) + ((v & 3) << 1);
}

__device__ inline float sigmoidf_(float x) { return 1.0f / (1.0f + __expf(-x)); }

__device__ inline v8f wmma_bf16(v8u a, v8u b, v8f acc) {
  return __builtin_amdgcn_wmma_f32_16x16x32_bf16(
      false, __builtin_bit_cast(v16bf, a),
      false, __builtin_bit_cast(v16bf, b),
      (short)0, acc, false, false);
}

// ---------------------------------------------------------------------------
// 1) ctx[t][r][c] = sum_hw x[r,hw,t,c] (r<8,c<512); gt appended; zero padded.
//    inp_f32 layout: [T][16 rows][KIN]  (rows 8..15 zero for the M=16 tile)
//    x is streamed once with NT hint so it does not evict the L2 working set
//    (packed weights + gi_all) that the serial recurrence re-reads 32x.
// ---------------------------------------------------------------------------
__global__ __launch_bounds__(256) void k_ctx(const float* __restrict__ x,
                                             const float* __restrict__ gt,
                                             float* __restrict__ inp) {
  int g = blockIdx.x * 256 + threadIdx.x;
  if (g >= T_ * 16 * KIN) return;
  int c = g % KIN;
  int r = (g / KIN) & 15;
  int t = g / (KIN * 16);
  float v = 0.0f;
  if (r < B_) {
    if (c < C_) {
      const float* p = x + ((size_t)r * HW_ * T_ + t) * C_ + c;
      float s = 0.0f;
#pragma unroll 4
      for (int hw = 0; hw < HW_; ++hw)
        s += __builtin_nontemporal_load(&p[(size_t)hw * T_ * C_]);
      v = s;
    } else if (c < C_ + OUT_) {
      v = gt[r * OUT_ + (c - C_)];
    }
  }
  inp[(t * 16 + r) * KIN + c] = v;
}

// ---------------------------------------------------------------------------
// 2) Pack inp into bf16 A-fragments: [T][KBI][lane 0..31][8 dwords]
// ---------------------------------------------------------------------------
__global__ __launch_bounds__(256) void k_pack_a(const float* __restrict__ inp,
                                                unsigned* __restrict__ frag) {
  int g = blockIdx.x * 256 + threadIdx.x;
  if (g >= T_ * KBI * 32) return;
  int lane = g & 31;
  int kb   = (g >> 5) % KBI;
  int t    = (g >> 5) / KBI;
  int row  = lane & 15;
  const float* src = inp + (t * 16 + row) * KIN + kb * 32;
  unsigned* dst = frag + (size_t)g * 8;
#pragma unroll
  for (int v = 0; v < 8; ++v) {
    int k = kpat(v, lane);
    dst[v] = pack_bf16(src[k], src[k + 1]);
  }
}

// ---------------------------------------------------------------------------
// 3) Pack W_ih^T (B = [516 x 1536], B[k][n] = W_ih[n][k]) into B-fragments:
//    [NB][KBI][lane][8 dwords]; lane holds column n = nb*16 + (lane&15).
// ---------------------------------------------------------------------------
__global__ __launch_bounds__(256) void k_pack_bih(const float* __restrict__ W,
                                                  unsigned* __restrict__ frag) {
  int g = blockIdx.x * 256 + threadIdx.x;
  if (g >= NB * KBI * 32) return;
  int lane = g & 31;
  int kb   = (g >> 5) % KBI;
  int nb   = (g >> 5) / KBI;
  int n    = nb * 16 + (lane & 15);
  unsigned* dst = frag + (size_t)g * 8;
#pragma unroll
  for (int v = 0; v < 8; ++v) {
    int k = kb * 32 + kpat(v, lane);
    float a = (k     < C_ + OUT_) ? W[(size_t)n * (C_ + OUT_) + k]     : 0.0f;
    float b = (k + 1 < C_ + OUT_) ? W[(size_t)n * (C_ + OUT_) + k + 1] : 0.0f;
    dst[v] = pack_bf16(a, b);
  }
}

// ---------------------------------------------------------------------------
// 4) Pack W_hh^T (B = [512 x 1536]) into B-fragments: [NB][KBH][lane][8 dw]
// ---------------------------------------------------------------------------
__global__ __launch_bounds__(256) void k_pack_bhh(const float* __restrict__ W,
                                                  unsigned* __restrict__ frag) {
  int g = blockIdx.x * 256 + threadIdx.x;
  if (g >= NB * KBH * 32) return;
  int lane = g & 31;
  int kb   = (g >> 5) & 15;
  int nb   = (g >> 5) >> 4;
  int n    = nb * 16 + (lane & 15);
  unsigned* dst = frag + (size_t)g * 8;
#pragma unroll
  for (int v = 0; v < 8; ++v) {
    int k = kb * 32 + kpat(v, lane);
    dst[v] = pack_bf16(W[(size_t)n * H_ + k], W[(size_t)n * H_ + k + 1]);
  }
}

// ---------------------------------------------------------------------------
// 5) gi_all[t] = inp[t] @ W_ih^T for ALL t in parallel (h-independent!).
//    One wave per 16x16 output tile; 32*96 = 3072 waves.
//    Software pipelined: fragments for kb+1 load while WMMA consumes kb.
// ---------------------------------------------------------------------------
__global__ __launch_bounds__(256) void k_gi(const unsigned* __restrict__ Afrag,
                                            const unsigned* __restrict__ Bfrag,
                                            float* __restrict__ gi) {
  int wave = blockIdx.x * 8 + (threadIdx.x >> 5);
  int lane = threadIdx.x & 31;
  if (wave >= T_ * NB) return;
  int t = wave / NB, nb = wave % NB;
  const v8u* ap = (const v8u*)(Afrag + ((size_t)t * KBI * 32 + lane) * 8);
  const v8u* bp = (const v8u*)(Bfrag + ((size_t)nb * KBI * 32 + lane) * 8);
  v8f acc = {};
  v8u a = ap[0], b = bp[0];
#pragma unroll 1
  for (int kb = 0; kb < KBI - 1; ++kb) {
    v8u an = ap[(kb + 1) * 32];
    v8u bn = bp[(kb + 1) * 32];
    acc = wmma_bf16(a, b, acc);
    a = an; b = bn;
  }
  acc = wmma_bf16(a, b, acc);
  int col = nb * 16 + (lane & 15);
  int rb  = (lane >> 4) * 8;
#pragma unroll
  for (int r = 0; r < 8; ++r)
    gi[((size_t)t * 16 + rb + r) * N3H + col] = acc[r];
}

// ---------------------------------------------------------------------------
// 6) Sequential GRU recurrence: single workgroup, 16 waves, LDS-resident.
//    Per step: pack h -> bf16 A-frags (LDS), 96 WMMA tiles of h @ W_hh^T
//    (B pipelined from L2, next tile prefetched), fused gates, capture h at
//    step ori_len[b]-1.
//    LDS: h 32KB | hA frags 16KB | gh 96KB = 144KB (of 320KB WGP pool).
// ---------------------------------------------------------------------------
__global__ __launch_bounds__(512) void k_gru(const float* __restrict__ gi,
                                             const unsigned* __restrict__ Bhh,
                                             const float* __restrict__ b_ih,
                                             const float* __restrict__ b_hh,
                                             const int* __restrict__ ori_len,
                                             float* __restrict__ hfin) {
  extern __shared__ char smem[];
  float*    hbuf = (float*)smem;                 // [16][512] f32
  unsigned* hA   = (unsigned*)(smem + 32768);    // [16 kb][32 lanes][8 dw]
  float*    gh   = (float*)(smem + 49152);       // [16][1536] f32

  int tid = threadIdx.x, lane = tid & 31, wave = tid >> 5;
  for (int i = tid; i < 16 * H_; i += 512) hbuf[i] = 0.0f;
  __syncthreads();

  for (int t = 0; t < T_; ++t) {
    // pack h into bf16 A-fragments (rows 8..15 are zero)
    {
      int kb = wave, row = lane & 15;
      const float* src = hbuf + row * H_ + kb * 32;
      unsigned* dst = hA + (size_t)(kb * 32 + lane) * 8;
#pragma unroll
      for (int v = 0; v < 8; ++v) {
        int k = kpat(v, lane);
        dst[v] = pack_bf16(src[k], src[k + 1]);
      }
    }
    __syncthreads();

    // gh = h @ W_hh^T : 96 tiles, 6 per wave; B from L2, pipelined
#pragma unroll 1
    for (int i = 0; i < 6; ++i) {
      int nb = wave + 16 * i;
      const v8u* bp = (const v8u*)(Bhh + ((size_t)nb * KBH * 32 + lane) * 8);
      if (i < 5) {  // prefetch next tile's B fragments (global_prefetch_b8)
        __builtin_prefetch(Bhh + ((size_t)(nb + 16) * KBH * 32 + lane) * 8, 0, 0);
      }
      v8f acc = {};
      v8u a = *(const v8u*)(hA + (size_t)lane * 8);
      v8u b = bp[0];
#pragma unroll 1
      for (int kb = 0; kb < KBH - 1; ++kb) {
        v8u an = *(const v8u*)(hA + (size_t)((kb + 1) * 32 + lane) * 8);
        v8u bn = bp[(kb + 1) * 32];
        acc = wmma_bf16(a, b, acc);
        a = an; b = bn;
      }
      acc = wmma_bf16(a, b, acc);
      int col = nb * 16 + (lane & 15);
      int rb  = (lane >> 4) * 8;
#pragma unroll
      for (int r = 0; r < 8; ++r) gh[(rb + r) * N3H + col] = acc[r];
    }
    __syncthreads();

    // fused GRU gates (torch GRUCell, gate order r,z,n)
    for (int e = tid; e < B_ * H_; e += 512) {
      int b = e >> 9, j = e & 511;
      const float* gib = gi + ((size_t)t * 16 + b) * N3H;
      float ir = gib[j]        + b_ih[j];
      float iz = gib[512 + j]  + b_ih[512 + j];
      float in = gib[1024 + j] + b_ih[1024 + j];
      float hr = gh[b * N3H + j]        + b_hh[j];
      float hz = gh[b * N3H + 512 + j]  + b_hh[512 + j];
      float hn = gh[b * N3H + 1024 + j] + b_hh[1024 + j];
      float r = sigmoidf_(ir + hr);
      float z = sigmoidf_(iz + hz);
      float n = tanhf(in + r * hn);
      float hnew = (1.0f - z) * n + z * hbuf[b * H_ + j];
      hbuf[b * H_ + j] = hnew;
      int idx = ori_len[b] - 1;
      idx = idx < 0 ? 0 : (idx > T_ - 1 ? T_ - 1 : idx);
      if (idx == t) hfin[b * H_ + j] = hnew;
    }
    __syncthreads();
  }
}

// ---------------------------------------------------------------------------
// 7) out = h_final @ Wfc + bfc  ([8,512] x [512,5] — trivially small)
// ---------------------------------------------------------------------------
__global__ __launch_bounds__(64) void k_fc(const float* __restrict__ hfin,
                                           const float* __restrict__ Wfc,
                                           const float* __restrict__ bfc,
                                           float* __restrict__ out) {
  int tid = threadIdx.x;
  if (tid >= B_ * (OUT_ + 1)) return;
  int b = tid / (OUT_ + 1), o = tid % (OUT_ + 1);
  float s = bfc[o];
  for (int j = 0; j < H_; ++j) s += hfin[b * H_ + j] * Wfc[j * (OUT_ + 1) + o];
  out[tid] = s;
}

// ---------------------------------------------------------------------------
extern "C" void kernel_launch(void* const* d_in, const int* in_sizes, int n_in,
                              void* d_out, int out_size, void* d_ws, size_t ws_size,
                              hipStream_t stream) {
  (void)in_sizes; (void)n_in; (void)out_size; (void)ws_size;
  const float* x       = (const float*)d_in[0];
  const int*   ori_len = (const int*)d_in[1];
  const float* gt      = (const float*)d_in[2];
  // d_in[3..8] = W1,b1,W2,b2,V,bV : provably dead (softmax over singleton == 1)
  const float* W_ih    = (const float*)d_in[9];
  const float* W_hh    = (const float*)d_in[10];
  const float* b_ih    = (const float*)d_in[11];
  const float* b_hh    = (const float*)d_in[12];
  const float* Wfc     = (const float*)d_in[13];
  const float* bfc     = (const float*)d_in[14];
  float* out = (float*)d_out;

  char* ws = (char*)d_ws;
  float*    inp_f32 = (float*)(ws + 0);          // 32*16*544*4      = 1,114,112
  unsigned* inpA    = (unsigned*)(ws + 1114112); // 32*17*32*32      =   557,056
  unsigned* Bih     = (unsigned*)(ws + 1671168); // 96*17*32*32      = 1,671,168
  unsigned* Bhh     = (unsigned*)(ws + 3342336); // 96*16*32*32      = 1,572,864
  float*    gi_all  = (float*)(ws + 4915200);    // 32*16*1536*4     = 3,145,728
  float*    hfin    = (float*)(ws + 8060928);    // 8*512*4          =    16,384

  k_ctx     <<<(T_ * 16 * KIN + 255) / 256, 256, 0, stream>>>(x, gt, inp_f32);
  k_pack_a  <<<(T_ * KBI * 32 + 255) / 256, 256, 0, stream>>>(inp_f32, inpA);
  k_pack_bih<<<(NB * KBI * 32 + 255) / 256, 256, 0, stream>>>(W_ih, Bih);
  k_pack_bhh<<<(NB * KBH * 32 + 255) / 256, 256, 0, stream>>>(W_hh, Bhh);
  k_gi      <<<(T_ * NB) / 8,               256, 0, stream>>>(inpA, Bih, gi_all);
  k_gru     <<<1, 512, 147456, stream>>>(gi_all, Bhh, b_ih, b_hh, ori_len, hfin);
  k_fc      <<<1, 64, 0, stream>>>(hfin, Wfc, bfc, out);
}
